// GraphCNN_5617817223311
// MI455X (gfx1250) — matrix operand order repaired and checked by reading, not compiled, not run on hardware
//
#include <hip/hip_runtime.h>
#include <hip/hip_bf16.h>

typedef __attribute__((ext_vector_type(16))) _Float16 v16h;
typedef __attribute__((ext_vector_type(8)))  _Float16 v8h;
typedef __attribute__((ext_vector_type(8)))  float    v8f;

#define N_NODES 131072
#define D_MODEL 128
#define NUM_LAYERS 4
#define NODES_PER_BLK 64

// ---------------------------------------------------------------------------
// Pack f32 weights [K x 128] into WMMA B-fragment order, f16:
//   dst[((s*8 + ntile)*32 + lane)*16 + j] = W[(s*32 + (lane>>4)*16 + j)*128 + ntile*16 + (lane&15)]
// B order used here (16-bit, 32x16; A matches): lanes 0-15 hold K=0..15 (col = lane), lanes 16-31 hold K=16..31.
// ---------------------------------------------------------------------------
__global__ __launch_bounds__(256) void pack_w_kernel(const float* __restrict__ W,
                                                     _Float16* __restrict__ Wf,
                                                     int ksteps, int layers) {
    int idx = blockIdx.x * blockDim.x + threadIdx.x;
    int total = layers * ksteps * 8 * 32 * 16;
    if (idx >= total) return;
    int j    = idx & 15;
    int lane = (idx >> 4) & 31;
    int t    = (idx >> 9) & 7;
    int s    = (idx >> 12) % ksteps;
    int l    = idx / (ksteps * 8 * 32 * 16);
    int K    = ksteps * 32;
    int k    = s * 32 + (lane >> 4) * 16 + j;
    int col  = t * 16 + (lane & 15);
    Wf[idx] = (_Float16)W[((long)l * K + k) * D_MODEL + col];
}

// ---------------------------------------------------------------------------
// Embedding lookup -> f16 features; block N_NODES zeroes the padding row of
// BOTH ping-pong buffers (gathers of the pad node must read zeros).
// ---------------------------------------------------------------------------
__global__ __launch_bounds__(128) void embed_kernel(const int* __restrict__ node_idx,
                                                    const float* __restrict__ emb,
                                                    _Float16* __restrict__ fA,
                                                    _Float16* __restrict__ fB) {
    int n = blockIdx.x;
    int d = threadIdx.x;
    if (n == N_NODES) {
        fA[(long)n * D_MODEL + d] = (_Float16)0.0f;
        fB[(long)n * D_MODEL + d] = (_Float16)0.0f;
        return;
    }
    fA[(long)n * D_MODEL + d] = (_Float16)emb[(long)node_idx[n] * D_MODEL + d];
}

// ---------------------------------------------------------------------------
// One GNN layer: gathered[64,512] @ W0 -> ReLU -> LDS -> @ W1 -> fout.
// 8 waves: wave w -> M-subtile (w&3), column half ((w>>2)*64).
// B (weights) are staged per-k-step into LDS cooperatively (8KB per k-step,
// double buffered) so all 8 waves share one L2 fetch and read via DS.
// ---------------------------------------------------------------------------
__global__ __launch_bounds__(256) void layer_kernel(const _Float16* __restrict__ fin,
                                                    _Float16* __restrict__ fout,
                                                    const int* __restrict__ neighbors,
                                                    const _Float16* __restrict__ W0f, // [16][8][32][16]
                                                    const float* __restrict__ b0,
                                                    const _Float16* __restrict__ W1f, // [4][8][32][16]
                                                    const float* __restrict__ b1) {
    __shared__ alignas(32) _Float16 hbuf[NODES_PER_BLK * D_MODEL]; // 16KB
    __shared__ alignas(32) _Float16 Bs[2][4096];                   // 2 x 8KB

    const int tid   = threadIdx.x;
    const int lane  = tid & 31;
    const int wave  = tid >> 5;
    const int m     = wave & 3;            // M-subtile (rows m*16 .. m*16+15)
    const int nh    = (wave >> 2) * 64;    // column half: 0 or 64
    const int nt0   = nh >> 4;             // first N-tile index (0 or 4)
    const int base  = blockIdx.x * NODES_PER_BLK;
    const int row16 = lane & 15;
    const int koff  = (lane >> 4) * 16;    // A layout: upper lanes take K+16..K+31
    const int node  = base + m * 16 + row16;

    // Hoist the 4 neighbor row pointers for this lane's matrix row.
    const _Float16* nrow[4];
#pragma unroll
    for (int nb = 0; nb < 4; ++nb) {
        nrow[nb] = fin + (long)neighbors[node * 4 + nb] * D_MODEL;
        __builtin_prefetch(nrow[nb], 0, 0);
    }

    union HU { v16h v; v8h h[2]; };

    // Cooperative stage of one 8KB k-step weight block into Bs[buf]:
    // 256 threads x 32B each (4096 halves total).
    auto stage = [&](const _Float16* __restrict__ Wf, int s, int buf) {
        const _Float16* src = Wf + (long)s * 4096 + tid * 16;
        v8h lo = *(const v8h*)(src);
        v8h hi = *(const v8h*)(src + 8);
        *(v8h*)(&Bs[buf][tid * 16])     = lo;
        *(v8h*)(&Bs[buf][tid * 16 + 8]) = hi;
    };

    // ---------------- GEMM 1: [64,512] x [512,128], K = 16 steps of 32 -----
    v8f acc[4] = {};
    stage(W0f, 0, 0);
    __syncthreads();
#pragma unroll
    for (int s = 0; s < 16; ++s) {
        const int cur = s & 1;
        if (s < 15) stage(W0f, s + 1, cur ^ 1);   // prefetch next k-step into other buffer
        // A fragment: direct gather from neighbor feature rows.
        const int nb = s >> 2;
        const int d0 = (s & 3) * 32 + koff;
        HU u; u.h[0] = *(const v8h*)(nrow[nb] + d0);
        u.h[1] = *(const v8h*)(nrow[nb] + d0 + 8);
        v16h a = u.v;
        // B fragments from LDS; 4 back-to-back identical WMMAs share A.
        v16h bt0 = *(const v16h*)(&Bs[cur][((nt0 + 0) * 32 + lane) * 16]);
        v16h bt1 = *(const v16h*)(&Bs[cur][((nt0 + 1) * 32 + lane) * 16]);
        v16h bt2 = *(const v16h*)(&Bs[cur][((nt0 + 2) * 32 + lane) * 16]);
        v16h bt3 = *(const v16h*)(&Bs[cur][((nt0 + 3) * 32 + lane) * 16]);
        acc[0] = __builtin_amdgcn_wmma_f32_16x16x32_f16(false, a, false, bt0, (short)0, acc[0], false, false);
        acc[1] = __builtin_amdgcn_wmma_f32_16x16x32_f16(false, a, false, bt1, (short)0, acc[1], true,  false);
        acc[2] = __builtin_amdgcn_wmma_f32_16x16x32_f16(false, a, false, bt2, (short)0, acc[2], true,  false);
        acc[3] = __builtin_amdgcn_wmma_f32_16x16x32_f16(false, a, false, bt3, (short)0, acc[3], true,  false);
        if (s < 15) __syncthreads();              // all reads of Bs[cur] done; stage of s+1 visible
    }

    // bias + ReLU + stash h (f16) in LDS for the second GEMM
#pragma unroll
    for (int t = 0; t < 4; ++t) {
        const int col   = nh + t * 16 + (lane & 15);
        const float bia = b0[col];
        const int rbase = m * 16 + (lane >> 4) * 8;
#pragma unroll
        for (int r = 0; r < 8; ++r) {
            float v = acc[t][r] + bia;
            v = v > 0.0f ? v : 0.0f;
            hbuf[(rbase + r) * D_MODEL + col] = (_Float16)v;
        }
    }
    __syncthreads();   // hbuf complete; also fences last Bs reads before GEMM2 staging

    // ---------------- GEMM 2: [64,128] x [128,128], K = 4 steps of 32 ------
    const _Float16* hrow = hbuf + (m * 16 + row16) * D_MODEL;
    v8f acc2[4] = {};
    stage(W1f, 0, 0);
    __syncthreads();
#pragma unroll
    for (int s = 0; s < 4; ++s) {
        const int cur = s & 1;
        if (s < 3) stage(W1f, s + 1, cur ^ 1);
        const int d0 = s * 32 + koff;
        HU u; u.h[0] = *(const v8h*)(hrow + d0);
        u.h[1] = *(const v8h*)(hrow + d0 + 8);
        v16h a = u.v;
        v16h bt0 = *(const v16h*)(&Bs[cur][((nt0 + 0) * 32 + lane) * 16]);
        v16h bt1 = *(const v16h*)(&Bs[cur][((nt0 + 1) * 32 + lane) * 16]);
        v16h bt2 = *(const v16h*)(&Bs[cur][((nt0 + 2) * 32 + lane) * 16]);
        v16h bt3 = *(const v16h*)(&Bs[cur][((nt0 + 3) * 32 + lane) * 16]);
        acc2[0] = __builtin_amdgcn_wmma_f32_16x16x32_f16(false, a, false, bt0, (short)0, acc2[0], false, false);
        acc2[1] = __builtin_amdgcn_wmma_f32_16x16x32_f16(false, a, false, bt1, (short)0, acc2[1], true,  false);
        acc2[2] = __builtin_amdgcn_wmma_f32_16x16x32_f16(false, a, false, bt2, (short)0, acc2[2], true,  false);
        acc2[3] = __builtin_amdgcn_wmma_f32_16x16x32_f16(false, a, false, bt3, (short)0, acc2[3], true,  false);
        if (s < 3) __syncthreads();
    }

#pragma unroll
    for (int t = 0; t < 4; ++t) {
        const int col   = nh + t * 16 + (lane & 15);
        const float bia = b1[col];
        const int rbase = m * 16 + (lane >> 4) * 8;
#pragma unroll
        for (int r = 0; r < 8; ++r) {
            float v = acc2[t][r] + bia;
            fout[(long)(base + rbase + r) * D_MODEL + col] = (_Float16)v;
        }
    }
}

// ---------------------------------------------------------------------------
// Readout: per-equation (128 nodes) segment means (64+64) and dot product.
// ---------------------------------------------------------------------------
__global__ __launch_bounds__(128) void readout_kernel(const _Float16* __restrict__ feats,
                                                      const float* __restrict__ out_bias,
                                                      float* __restrict__ out) {
    __shared__ float red[128];
    const int b = blockIdx.x;
    const int d = threadIdx.x;
    const _Float16* base = feats + (long)b * 128 * D_MODEL;
    float sl = 0.0f, sr = 0.0f;
#pragma unroll 4
    for (int i = 0; i < 64; ++i) {
        sl += (float)base[i * D_MODEL + d];
        sr += (float)base[(64 + i) * D_MODEL + d];
    }
    float p = (sl * (1.0f / 64.0f)) * (sr * (1.0f / 64.0f));
    red[d] = p;
    __syncthreads();
    for (int s = 64; s > 0; s >>= 1) {
        if (d < s) red[d] += red[d + s];
        __syncthreads();
    }
    if (d == 0) out[b] = red[0] + out_bias[0];
}

// ---------------------------------------------------------------------------
extern "C" void kernel_launch(void* const* d_in, const int* in_sizes, int n_in,
                              void* d_out, int out_size, void* d_ws, size_t ws_size,
                              hipStream_t stream) {
    const int*   node_idx  = (const int*)  d_in[0];   // [N]
    const int*   neighbors = (const int*)  d_in[1];   // [N,4]
    // d_in[2] = seg_ids (implicit contiguous layout; unused)
    const float* emb       = (const float*)d_in[3];   // [320,128]
    const float* W0        = (const float*)d_in[4];   // [4,512,128]
    const float* b0        = (const float*)d_in[5];   // [4,128]
    const float* W1        = (const float*)d_in[6];   // [4,128,128]
    const float* b1        = (const float*)d_in[7];   // [4,128]
    const float* out_bias  = (const float*)d_in[8];   // [1]
    float*       out       = (float*)d_out;           // [1024]

    // Workspace layout (halves); all sub-buffers stay 256-byte aligned.
    const long featsElems = (long)(N_NODES + 1) * D_MODEL;  // includes padding row
    _Float16* fA  = (_Float16*)d_ws;
    _Float16* fB  = fA + featsElems;
    _Float16* W0f = fB + featsElems;                         // 4*16*8*32*16 halves
    _Float16* W1f = W0f + (long)NUM_LAYERS * 16 * 8 * 32 * 16;

    // Pack weights into WMMA B-fragment order (f16).
    {
        int total0 = NUM_LAYERS * 16 * 8 * 32 * 16;
        pack_w_kernel<<<(total0 + 255) / 256, 256, 0, stream>>>(W0, W0f, 16, NUM_LAYERS);
        int total1 = NUM_LAYERS * 4 * 8 * 32 * 16;
        pack_w_kernel<<<(total1 + 255) / 256, 256, 0, stream>>>(W1, W1f, 4, NUM_LAYERS);
    }

    // Embedding lookup (+ zero the padding rows of both ping-pong buffers).
    embed_kernel<<<N_NODES + 1, 128, 0, stream>>>(node_idx, emb, fA, fB);

    // 4 message-passing layers, ping-pong fA <-> fB.
    _Float16* cur = fA;
    _Float16* nxt = fB;
    for (int l = 0; l < NUM_LAYERS; ++l) {
        layer_kernel<<<N_NODES / NODES_PER_BLK, 256, 0, stream>>>(
            cur, nxt, neighbors,
            W0f + (long)l * 16 * 8 * 32 * 16, b0 + l * D_MODEL,
            W1f + (long)l * 4  * 8 * 32 * 16, b1 + l * D_MODEL);
        _Float16* t = cur; cur = nxt; nxt = t;
    }
    // After 4 layers the result lives in `cur` (== fA).

    // Segment means + per-equation dot product.
    readout_kernel<<<1024, 128, 0, stream>>>(cur, out_bias, out);
}
